// SetAbstraction_68281390072561
// MI455X (gfx1250) — compile-verified, matrix operand order checked by reading, not run on hardware
//
#include <hip/hip_runtime.h>

typedef __attribute__((ext_vector_type(16))) _Float16 v16h;
typedef __attribute__((ext_vector_type(8)))  _Float16 v8h;
typedef __attribute__((ext_vector_type(8)))  float    v8f;

#define B_      8
#define N_      4096
#define CIN_    64
#define COUT_   128
#define K_      16
#define NPOINT_ 1024
#define P_      (B_ * N_)                 // 32768 points
#define M_      ((size_t)P_ * K_)         // 524288 edge rows
#define EPS_    1e-5f

__device__ __forceinline__ v8f wmma16x16x32(v16h a, v16h b, v8f c) {
  return __builtin_amdgcn_wmma_f32_16x16x32_f16(false, a, false, b, (short)0, c,
                                                false, false);
}

// Build a 16-element A fragment from a contiguous f16 row.
// col(e) = colbase + (e>>3)*16 + half*8 + (e&7)
__device__ __forceinline__ v16h loadA16(const _Float16* row, int colbase, int half) {
  v8h lo = *(const v8h*)(row + colbase + half * 8);
  v8h hi = *(const v8h*)(row + colbase + 16 + half * 8);
  v16h r;
#pragma unroll
  for (int e = 0; e < 8; ++e) { r[e] = lo[e]; r[e + 8] = hi[e]; }
  return r;
}

// ---------------------------------------------------------------- prep kernels
__global__ void prep_featT(const float* feat, _Float16* featT) {
  size_t id = (size_t)blockIdx.x * blockDim.x + threadIdx.x;   // B*CIN*N
  int b = (int)(id / (CIN_ * N_));
  int r = (int)(id % (CIN_ * N_));
  int c = r / N_, n = r % N_;
  featT[((size_t)b * N_ + n) * CIN_ + c] = (_Float16)feat[id];
}

__global__ void prep_xyzT(const float* xyz, float* xyzT) {
  size_t id = (size_t)blockIdx.x * blockDim.x + threadIdx.x;   // B*3*N
  int b = (int)(id / (3 * N_));
  int r = (int)(id % (3 * N_));
  int d = r / N_, n = r % N_;
  xyzT[((size_t)b * N_ + n) * 3 + d] = xyz[id];
}

// Pack weight [Krows x Ncols] (row-major f32) into per-lane WMMA B fragments.
// out index = ((s*Jtiles + j)*32 + lane)*16 + e
__global__ void pack_w(const float* w, int Krows, int Ncols, int Jtiles,
                       int total, _Float16* out) {
  int id = blockIdx.x * blockDim.x + threadIdx.x;
  if (id >= total) return;
  int e    = id & 15;
  int lane = (id >> 4) & 31;
  int j    = (id >> 9) % Jtiles;
  int s    = id / (512 * Jtiles);
  int col  = j * 16 + (lane & 15);
  int krow = s * 32 + (e >> 3) * 16 + (lane >> 4) * 8 + (e & 7);
  float v = (krow < Krows) ? w[(size_t)krow * Ncols + col] : 0.0f;
  out[id] = (_Float16)v;
}

__global__ void zero_f32(float* p, int n) {
  int i = blockIdx.x * blockDim.x + threadIdx.x;
  if (i < n) p[i] = 0.0f;
}

// ---------------------------------------------------------------- KNN (top-16)
__global__ __launch_bounds__(256) void knn_kernel(const float* xyz, int* knn) {
  __shared__ float sx[256], sy[256], sz[256];
  int b  = blockIdx.x / (N_ / 256);
  int n0 = (blockIdx.x % (N_ / 256)) * 256;
  int tid = threadIdx.x;
  int n = n0 + tid;
  const float* xb = xyz + (size_t)b * 3 * N_;
  float qx = xb[n], qy = xb[N_ + n], qz = xb[2 * N_ + n];
  float bd[K_]; int bi[K_];
#pragma unroll
  for (int i = 0; i < K_; ++i) { bd[i] = 1e30f; bi[i] = 0; }
  for (int t = 0; t < N_; t += 256) {
    __syncthreads();
    sx[tid] = xb[t + tid]; sy[tid] = xb[N_ + t + tid]; sz[tid] = xb[2 * N_ + t + tid];
    __syncthreads();
    for (int m = 0; m < 256; ++m) {
      float dx = sx[m] - qx, dy = sy[m] - qy, dz = sz[m] - qz;
      float d = dx * dx + dy * dy + dz * dz;
      if (d < bd[K_ - 1]) {
        int pos = K_ - 1;
#pragma unroll
        for (int i = K_ - 2; i >= 0; --i)
          if (d < bd[i]) { bd[i + 1] = bd[i]; bi[i + 1] = bi[i]; pos = i; }
        bd[pos] = d; bi[pos] = t + m;
      }
    }
  }
  size_t q = (size_t)b * N_ + n;
#pragma unroll
  for (int i = 0; i < K_; ++i) knn[q * K_ + i] = bi[i];
}

// ------------------------------------------------------- GEMM1: edge * w1 (WMMA)
// One wave per point: 16 rows (K neighbors) x 128 channels, Kdim 131 -> 5 steps.
__global__ __launch_bounds__(256) void gemm1_kernel(const _Float16* featT,
                                                    const float* xyzT,
                                                    const int* knn,
                                                    const _Float16* wpack,
                                                    float* y1) {
  int wave = (int)(((size_t)blockIdx.x * blockDim.x + threadIdx.x) >> 5);
  int lane = threadIdx.x & 31;
  int p = wave;
  int b = p >> 12;
  int half = lane >> 4, cr = lane & 15;
  const _Float16* ctrRow = featT + (size_t)p * CIN_;
  int idxv = knn[(size_t)p * K_ + cr];
  const _Float16* nbRow = featT + ((size_t)b * N_ + idxv) * CIN_;

  v8f acc[8];
#pragma unroll
  for (int j = 0; j < 8; ++j) acc[j] = {};

#pragma unroll
  for (int s = 0; s < 5; ++s) {
    v16h a;
    if (s < 2) {
      a = loadA16(ctrRow, s * 32, half);
    } else if (s < 4) {
      a = loadA16(nbRow, (s - 2) * 32, half);
    } else {
      v16h z;
#pragma unroll
      for (int e = 0; e < 16; ++e) z[e] = (_Float16)0.0f;
      if (half == 0) {   // cols 128..130 = rel xyz
        const float* q  = xyzT + (size_t)p * 3;
        const float* nb = xyzT + ((size_t)b * N_ + idxv) * 3;
        z[0] = (_Float16)(nb[0] - q[0]);
        z[1] = (_Float16)(nb[1] - q[1]);
        z[2] = (_Float16)(nb[2] - q[2]);
      }
      a = z;
    }
#pragma unroll
    for (int j = 0; j < 8; ++j) {
      v16h bf = *(const v16h*)(wpack + (((s * 8 + j) * 32) + lane) * 16);
      acc[j] = wmma16x16x32(a, bf, acc[j]);
    }
  }
  float* outb = y1 + ((size_t)p * 16 + 8 * half) * COUT_;
#pragma unroll
  for (int j = 0; j < 8; ++j) {
    int c = j * 16 + cr;
#pragma unroll
    for (int v = 0; v < 8; ++v) outb[(size_t)v * COUT_ + c] = acc[j][v];
  }
}

// -------------------------------------- generic WMMA GEMM with fused BN+ReLU on A
// X: [M][128] f32, applies relu(x*scale+shift) -> f16 A rows; B from wpack.
template <int KS, int JT, int COUTK>
__global__ __launch_bounds__(256) void gemm_bn_kernel(const float* X,
                                                      const float* scale,
                                                      const float* shift,
                                                      const _Float16* wpack,
                                                      float* Y) {
  __shared__ float ssc[128], ssh[128];
  int tid = threadIdx.x;
  if (tid < 128) { ssc[tid] = scale[tid]; ssh[tid] = shift[tid]; }
  __syncthreads();
  int wave = (int)(((size_t)blockIdx.x * blockDim.x + tid) >> 5);
  int lane = tid & 31;
  int half = lane >> 4, cr = lane & 15;
  const float* rp = X + ((size_t)wave * 16 + cr) * 128;

  v8f acc[JT];
#pragma unroll
  for (int j = 0; j < JT; ++j) acc[j] = {};

#pragma unroll
  for (int s = 0; s < KS; ++s) {
    const float* q = rp + s * 32 + half * 8;
    v16h a;
#pragma unroll
    for (int e = 0; e < 8; ++e) {
      int c0 = s * 32 + half * 8 + e;
      int c1 = c0 + 16;
      float v0 = q[e] * ssc[c0] + ssh[c0];      v0 = v0 > 0.f ? v0 : 0.f;
      float v1 = q[16 + e] * ssc[c1] + ssh[c1]; v1 = v1 > 0.f ? v1 : 0.f;
      a[e] = (_Float16)v0; a[e + 8] = (_Float16)v1;
    }
#pragma unroll
    for (int j = 0; j < JT; ++j) {
      v16h bf = *(const v16h*)(wpack + (((s * JT + j) * 32) + lane) * 16);
      acc[j] = wmma16x16x32(a, bf, acc[j]);
    }
  }
  float* outb = Y + ((size_t)wave * 16 + 8 * half) * COUTK;
#pragma unroll
  for (int j = 0; j < JT; ++j) {
    int c = j * 16 + cr;
#pragma unroll
    for (int v = 0; v < 8; ++v) outb[(size_t)v * COUTK + c] = acc[j][v];
  }
}

// ----------------------------------------------------------- per-channel stats
template <int C>
__global__ __launch_bounds__(256) void stats_kernel(const float* X, size_t Mrows,
                                                    float* sums) {  // sums[2C]
  __shared__ float ss[128], sq[128];
  int tid = threadIdx.x;
  if (tid < C) { ss[tid] = 0.f; sq[tid] = 0.f; }
  __syncthreads();
  size_t total = Mrows * (size_t)C;
  for (size_t i = (size_t)blockIdx.x * blockDim.x + tid; i < total;
       i += (size_t)gridDim.x * blockDim.x) {
    float v = X[i];
    int c = (int)(i & (size_t)(C - 1));
    atomicAdd(&ss[c], v);
    atomicAdd(&sq[c], v * v);
  }
  __syncthreads();
  if (tid < C) { atomicAdd(&sums[tid], ss[tid]); atomicAdd(&sums[C + tid], sq[tid]); }
}

__global__ void finalize_stats(const float* sums, int C, float invM,
                               const float* gamma, const float* beta,
                               float* scale, float* shift) {
  int c = blockIdx.x * blockDim.x + threadIdx.x;
  if (c < C) {
    float mean = sums[c] * invM;
    float var  = sums[C + c] * invM - mean * mean;
    float sc   = gamma[c] * rsqrtf(var + EPS_);
    scale[c] = sc;
    shift[c] = beta[c] - mean * sc;
  }
}

// ------------------------------------------- attention softmax + weighted pool
__global__ __launch_bounds__(128) void pool_kernel(const float* y2,
                                                   const float* s2, const float* sh2,
                                                   const float* h,
                                                   const float* sa, const float* sha,
                                                   const float* wa2, float* outpool) {
  __shared__ float logit[16];
  __shared__ float lsc[128], lsh[128];
  int p = blockIdx.x, tid = threadIdx.x;
  lsc[tid] = s2[tid]; lsh[tid] = sh2[tid];
  if (tid < 16) {
    const float* hr = h + ((size_t)p * 16 + tid) * 32;
    float acc = 0.f;
    for (int c = 0; c < 32; ++c) {
      float v = hr[c] * sa[c] + sha[c];
      v = v > 0.f ? v : 0.f;
      acc += v * wa2[c];
    }
    logit[tid] = acc;
  }
  __syncthreads();
  float mx = -1e30f;
#pragma unroll
  for (int k = 0; k < 16; ++k) mx = fmaxf(mx, logit[k]);
  float se = 0.f;
#pragma unroll
  for (int k = 0; k < 16; ++k) se += __expf(logit[k] - mx);
  const float* yb = y2 + (size_t)p * 16 * 128 + tid;
  float o = 0.f;
#pragma unroll
  for (int k = 0; k < 16; ++k) {
    float v = yb[(size_t)k * 128] * lsc[tid] + lsh[tid];
    v = v > 0.f ? v : 0.f;
    o += v * __expf(logit[k] - mx);
  }
  outpool[(size_t)p * 128 + tid] = o / se;
}

// -------------------------------------------------- farthest point sampling
__global__ __launch_bounds__(256) void fps_kernel(const float* xyz, int* fps_idx) {
  __shared__ float dist[N_];
  __shared__ float rv[256];
  __shared__ int   ri[256];
  __shared__ int   sfar;
  int b = blockIdx.x, tid = threadIdx.x;
  const float* xb = xyz + (size_t)b * 3 * N_;
  for (int i = tid; i < N_; i += 256) dist[i] = 1e10f;
  if (tid == 0) sfar = 0;
  __syncthreads();
  for (int it = 0; it < NPOINT_; ++it) {
    int far = sfar;
    if (tid == 0) fps_idx[b * NPOINT_ + it] = far;
    float cx = xb[far], cy = xb[N_ + far], cz = xb[2 * N_ + far];
    float bv = -1.f; int bidx = 0;
    for (int i = tid; i < N_; i += 256) {
      float dx = xb[i] - cx, dy = xb[N_ + i] - cy, dz = xb[2 * N_ + i] - cz;
      float d = dx * dx + dy * dy + dz * dz;
      float dm = fminf(dist[i], d);
      dist[i] = dm;
      if (dm > bv) { bv = dm; bidx = i; }
    }
    rv[tid] = bv; ri[tid] = bidx;
    __syncthreads();
    for (int s = 128; s > 0; s >>= 1) {
      if (tid < s) {
        if (rv[tid + s] > rv[tid] ||
            (rv[tid + s] == rv[tid] && ri[tid + s] < ri[tid])) {
          rv[tid] = rv[tid + s]; ri[tid] = ri[tid + s];
        }
      }
      __syncthreads();
    }
    if (tid == 0) sfar = ri[0];
    __syncthreads();
  }
}

__global__ void gather_kernel(const float* outpool, const int* fps_idx, float* gbuf) {
  size_t id = (size_t)blockIdx.x * blockDim.x + threadIdx.x;  // B*NPOINT*128
  int c = (int)(id & 127);
  size_t bp = id >> 7;
  int pnt = (int)(bp & (NPOINT_ - 1));
  int b = (int)(bp >> 10);
  int src = fps_idx[b * NPOINT_ + pnt];
  gbuf[id] = outpool[((size_t)b * N_ + src) * 128 + c];
}

__global__ void final_write(const float* gbuf, const float* scf, const float* shf,
                            const float* xyz, const int* fps_idx, float* out,
                            int total) {
  int id = blockIdx.x * blockDim.x + threadIdx.x;
  if (id >= total) return;
  const int XYZ_SZ = B_ * 3 * NPOINT_;                 // 24576
  if (id < XYZ_SZ) {
    int pnt = id % NPOINT_;
    int d = (id / NPOINT_) % 3;
    int b = id / (3 * NPOINT_);
    int src = fps_idx[b * NPOINT_ + pnt];
    out[id] = xyz[((size_t)b * 3 + d) * N_ + src];
  } else {
    size_t k = (size_t)(id - XYZ_SZ);                  // b*128*1024 layout
    int pnt = (int)(k & (NPOINT_ - 1));
    size_t t = k >> 10;
    int c = (int)(t & 127);
    int b = (int)(t >> 7);
    float v = gbuf[((size_t)b * NPOINT_ + pnt) * 128 + c];
    out[id] = v * scf[c] + shf[c];
  }
}

// ------------------------------------------------------------------- launcher
extern "C" void kernel_launch(void* const* d_in, const int* in_sizes, int n_in,
                              void* d_out, int out_size, void* d_ws, size_t ws_size,
                              hipStream_t stream) {
  const float* xyz  = (const float*)d_in[0];
  const float* feat = (const float*)d_in[1];
  const float* w1   = (const float*)d_in[2];
  const float* g1   = (const float*)d_in[4];
  const float* be1  = (const float*)d_in[5];
  const float* w2   = (const float*)d_in[6];
  const float* g2   = (const float*)d_in[8];
  const float* be2  = (const float*)d_in[9];
  const float* wa1  = (const float*)d_in[10];
  const float* ga   = (const float*)d_in[12];
  const float* bea  = (const float*)d_in[13];
  const float* wa2  = (const float*)d_in[14];
  const float* gf   = (const float*)d_in[16];
  const float* bf   = (const float*)d_in[17];
  float* out = (float*)d_out;

  char* ws = (char*)d_ws;
  size_t off = 0;
  auto alloc = [&](size_t bytes) {
    size_t o = off;
    off += (bytes + 255) & ~(size_t)255;
    return o;
  };
  size_t o_featT  = alloc((size_t)P_ * CIN_ * 2);
  size_t o_xyzT   = alloc((size_t)P_ * 3 * 4);
  size_t o_knn    = alloc((size_t)P_ * K_ * 4);
  size_t o_wp1    = alloc(5 * 8 * 512 * 2);
  size_t o_wp2    = alloc(4 * 8 * 512 * 2);
  size_t o_wpa    = alloc(4 * 2 * 512 * 2);
  size_t o_sums   = alloc(832 * 4);       // y1:256 | y2:256 | h:64 | g:256
  size_t o_ss     = alloc(832 * 4);       // s1,sh1,s2,sh2 (4*128) | sa,sha (2*32) | sf,shf (2*128)
  size_t o_y1     = alloc(M_ * 128 * 4);  // h [M][32] aliases this region
  size_t o_y2     = alloc(M_ * 128 * 4);  // gather buf aliases this region
  size_t o_pool   = alloc((size_t)P_ * 128 * 4);
  size_t o_fps    = alloc((size_t)B_ * NPOINT_ * 4);
  (void)ws_size; (void)n_in; (void)in_sizes; (void)out_size;

  _Float16* featT = (_Float16*)(ws + o_featT);
  float*    xyzT  = (float*)(ws + o_xyzT);
  int*      knn   = (int*)(ws + o_knn);
  _Float16* wp1   = (_Float16*)(ws + o_wp1);
  _Float16* wp2   = (_Float16*)(ws + o_wp2);
  _Float16* wpa   = (_Float16*)(ws + o_wpa);
  float* sums_y1 = (float*)(ws + o_sums);
  float* sums_y2 = sums_y1 + 256;
  float* sums_h  = sums_y1 + 512;
  float* sums_g  = sums_y1 + 576;
  float* s1  = (float*)(ws + o_ss);
  float* sh1 = s1 + 128;
  float* s2  = s1 + 256;
  float* sh2 = s1 + 384;
  float* sa  = s1 + 512;
  float* sha = s1 + 544;
  float* sf  = s1 + 576;
  float* shf = s1 + 704;
  float* y1   = (float*)(ws + o_y1);
  float* h    = y1;                       // alias (y1 dead after gemm2)
  float* y2   = (float*)(ws + o_y2);
  float* gbuf = y2;                       // alias (y2 dead after pool)
  float* pool = (float*)(ws + o_pool);
  int*   fpsI = (int*)(ws + o_fps);

  // 0) zero stat accumulators
  zero_f32<<<4, 256, 0, stream>>>(sums_y1, 832);

  // 1) prep transposed inputs + packed weights
  prep_featT<<<(P_ * CIN_) / 256, 256, 0, stream>>>(feat, featT);
  prep_xyzT<<<(P_ * 3) / 256, 256, 0, stream>>>(xyz, xyzT);
  pack_w<<<(5 * 8 * 512 + 255) / 256, 256, 0, stream>>>(w1, 131, 128, 8, 5 * 8 * 512, wp1);
  pack_w<<<(4 * 8 * 512 + 255) / 256, 256, 0, stream>>>(w2, 128, 128, 8, 4 * 8 * 512, wp2);
  pack_w<<<(4 * 2 * 512 + 255) / 256, 256, 0, stream>>>(wa1, 128, 32, 2, 4 * 2 * 512, wpa);

  // 2) KNN
  knn_kernel<<<P_ / 256, 256, 0, stream>>>(xyz, knn);

  // 3) edge GEMM1 (WMMA), global BN1 stats, finalize
  gemm1_kernel<<<P_ / 8, 256, 0, stream>>>(featT, xyzT, knn, wp1, y1);
  stats_kernel<128><<<1024, 256, 0, stream>>>(y1, M_, sums_y1);
  finalize_stats<<<1, 128, 0, stream>>>(sums_y1, 128, 1.0f / (float)M_, g1, be1, s1, sh1);

  // 4) GEMM2 (fused BN1+ReLU on A), BN2 stats
  gemm_bn_kernel<4, 8, 128><<<P_ / 8, 256, 0, stream>>>(y1, s1, sh1, wp2, y2);
  stats_kernel<128><<<1024, 256, 0, stream>>>(y2, M_, sums_y2);
  finalize_stats<<<1, 128, 0, stream>>>(sums_y2, 128, 1.0f / (float)M_, g2, be2, s2, sh2);

  // 5) attention hidden (fused BN2+ReLU on A), BN-a stats
  gemm_bn_kernel<4, 2, 32><<<P_ / 8, 256, 0, stream>>>(y2, s2, sh2, wpa, h);
  stats_kernel<32><<<1024, 256, 0, stream>>>(h, M_, sums_h);
  finalize_stats<<<1, 32, 0, stream>>>(sums_h, 32, 1.0f / (float)M_, ga, bea, sa, sha);

  // 6) softmax attention pooling
  pool_kernel<<<P_, 128, 0, stream>>>(y2, s2, sh2, h, sa, sha, wa2, pool);

  // 7) FPS + gather + final BN
  fps_kernel<<<B_, 256, 0, stream>>>(xyz, fpsI);
  gather_kernel<<<(B_ * NPOINT_ * 128) / 256, 256, 0, stream>>>(pool, fpsI, gbuf);
  stats_kernel<128><<<256, 256, 0, stream>>>(gbuf, (size_t)B_ * NPOINT_, sums_g);
  finalize_stats<<<1, 128, 0, stream>>>(sums_g, 128, 1.0f / (float)(B_ * NPOINT_),
                                        gf, bf, sf, shf);

  int total = B_ * 3 * NPOINT_ + B_ * COUT_ * NPOINT_;
  final_write<<<(total + 255) / 256, 256, 0, stream>>>(gbuf, sf, shf, xyz, fpsI, out, total);
}
